// SelfAttention_60979945668751
// MI455X (gfx1250) — matrix operand order repaired and checked by reading, not compiled, not run on hardware
//
#include <hip/hip_runtime.h>
#include <hip/hip_bf16.h>

typedef __attribute__((ext_vector_type(16))) _Float16 v16h;
typedef __attribute__((ext_vector_type(8)))  _Float16 v8h;
typedef __attribute__((ext_vector_type(8)))  float    v8f;

#define WMMA_F16(a,b,c) __builtin_amdgcn_wmma_f32_16x16x32_f16( \
    false, (a), false, (b), (short)0, (c), false, false)

static constexpr int EMB  = 1024;
static constexpr int NH   = 16;
static constexpr int HD   = 64;
static constexpr int BSZ  = 2;
static constexpr int SEQ  = 2048;
static constexpr int TOK  = BSZ * SEQ;   // 4096

__device__ inline v8f zero8() {
  v8f z;
#pragma unroll
  for (int i = 0; i < 8; ++i) z[i] = 0.0f;
  return z;
}

// -----------------------------------------------------------------------------
// Kernel 0: one-shot f32 -> f16 conversion (x and weight matrices).
// Removes all v_cvt work from the GEMM hot loops and halves streamed bytes.
// -----------------------------------------------------------------------------
__global__ __launch_bounds__(256) void cvt_f16_kernel(
    const float* __restrict__ src, _Float16* __restrict__ dst, int n) {
  const int i = (blockIdx.x * 256 + threadIdx.x) * 8;
  if (i + 8 <= n) {
    const float4 a = *(const float4*)(src + i);
    const float4 b = *(const float4*)(src + i + 4);
    v8h h;
    h[0] = (_Float16)a.x; h[1] = (_Float16)a.y;
    h[2] = (_Float16)a.z; h[3] = (_Float16)a.w;
    h[4] = (_Float16)b.x; h[5] = (_Float16)b.y;
    h[6] = (_Float16)b.z; h[7] = (_Float16)b.w;
    *(v8h*)(dst + i) = h;
  }
}

// -----------------------------------------------------------------------------
// Kernel 1: fused QKV projection.  y = x @ W^T + b  (torch Linear semantics)
// All operands pre-converted f16; inner loop is b128 loads + WMMA only.
// Q,K written [B*H, S, D] f16 row-major; V written transposed [B*H, D, S].
// -----------------------------------------------------------------------------
__global__ __launch_bounds__(128) void qkv_proj_kernel(
    const _Float16* __restrict__ xh,
    const _Float16* __restrict__ Wqh, const float* __restrict__ bq,
    const _Float16* __restrict__ Wkh, const float* __restrict__ bk,
    const _Float16* __restrict__ Wvh, const float* __restrict__ bv,
    _Float16* __restrict__ Qb, _Float16* __restrict__ Kb,
    _Float16* __restrict__ Vt) {
  const int lane = threadIdx.x & 31;
  const int wave = threadIdx.x >> 5;
  const int l16  = lane & 15;
  const int half = (lane >> 4) & 1;
  const int m0 = (blockIdx.x * 4 + wave) * 16;   // token tile
  const int n0 = blockIdx.y * 64;                // feature tile
  const int which = blockIdx.z;                  // 0=Q 1=K 2=V
  const _Float16* W = (which == 0) ? Wqh : (which == 1) ? Wkh : Wvh;
  const float* bias = (which == 0) ? bq : (which == 1) ? bk : bv;

  v8f acc[4];
#pragma unroll
  for (int t = 0; t < 4; ++t) acc[t] = zero8();

  const _Float16* xrow = xh + (size_t)(m0 + l16) * EMB;
  const _Float16* wrow[4];
#pragma unroll
  for (int t = 0; t < 4; ++t)
    wrow[t] = W + (size_t)(n0 + 16 * t + l16) * EMB + half * 16;

  for (int k0 = 0; k0 < EMB; k0 += 32) {
    // A fragment: lane = row; two 8-half chunks, k = 16*half + i (same order as B)
    v8h a0 = *(const v8h*)(xrow + k0 + half * 16);
    v8h a1 = *(const v8h*)(xrow + k0 + half * 16 + 8);
    v16h a;
#pragma unroll
    for (int i = 0; i < 8; ++i) { a[i] = a0[i]; a[8 + i] = a1[i]; }
#pragma unroll
    for (int t = 0; t < 4; ++t) {
      // B fragment: B[k][n] = W[n][k]; lane = col n, 16 contiguous k
      v16h bf = *(const v16h*)(wrow[t] + k0);
      __builtin_prefetch((const void*)(wrow[t] + k0 + 128), 0, 0);
      acc[t] = WMMA_F16(a, bf, acc[t]);
    }
  }

#pragma unroll
  for (int t = 0; t < 4; ++t) {
    const int f = n0 + 16 * t + l16;      // output feature (lane column)
    const int head = f >> 6;
    const int d    = f & 63;
    const float bb = bias[f];
    if (which < 2) {
      _Float16* dst = (which == 0) ? Qb : Kb;
#pragma unroll
      for (int r = 0; r < 8; ++r) {
        const int tok = m0 + 8 * half + r;        // C-frag row -> token
        const int b = tok >> 11, s = tok & (SEQ - 1);
        dst[(((size_t)(b * NH + head)) * SEQ + s) * HD + d] =
            (_Float16)(acc[t][r] + bb);
      }
    } else {
      // transposed V: consecutive rows -> consecutive s -> one b128 store
      const int tok0 = m0 + 8 * half;
      const int b = tok0 >> 11, s0 = tok0 & (SEQ - 1);
      v8h hv;
#pragma unroll
      for (int r = 0; r < 8; ++r) hv[r] = (_Float16)(acc[t][r] + bb);
      *(v8h*)(Vt + (((size_t)(b * NH + head)) * HD + d) * SEQ + s0) = hv;
    }
  }
}

// -----------------------------------------------------------------------------
// Kernel 2: causal flash attention.  Block = 64 queries of one (b,h);
// each wave = 16 queries.  Online softmax; scores never touch memory.
// -----------------------------------------------------------------------------
__global__ __launch_bounds__(128) void attn_kernel(
    const _Float16* __restrict__ Qb, const _Float16* __restrict__ Kb,
    const _Float16* __restrict__ Vt, _Float16* __restrict__ attn) {
  __shared__ _Float16 pbuf[4][16][32];   // per-wave P re-layout buffer (4KB)

  const int lane = threadIdx.x & 31;
  const int wave = threadIdx.x >> 5;
  const int l16  = lane & 15;
  const int half = (lane >> 4) & 1;
  const int bh = blockIdx.y;             // 0..31
  const int q0 = blockIdx.x * 64;
  const int qw = q0 + wave * 16;

  // Q A-fragments (head dim 64 = 2 K-slices of 32)
  const _Float16* qrow = Qb + ((size_t)bh * SEQ + qw + l16) * HD;
  v16h qa[2];
#pragma unroll
  for (int kk = 0; kk < 2; ++kk) {
    v8h c0 = *(const v8h*)(qrow + kk * 32 + half * 16);
    v8h c1 = *(const v8h*)(qrow + kk * 32 + half * 16 + 8);
#pragma unroll
    for (int i = 0; i < 8; ++i) { qa[kk][i] = c0[i]; qa[kk][8 + i] = c1[i]; }
  }

  v8f o[4];
#pragma unroll
  for (int t = 0; t < 4; ++t) o[t] = zero8();
  float rm[8], rl[8];
#pragma unroll
  for (int r = 0; r < 8; ++r) { rm[r] = -__builtin_inff(); rl[r] = 0.0f; }

  const float scale = 0.125f;            // 1/sqrt(64)
  const int nj = 2 * blockIdx.x + 2;     // 32-key blocks covering causal span

  for (int j = 0; j < nj; ++j) {
    const int kb0 = j * 32;
    // ---- scores S[16q x 32k] = Q @ K^T : 4 WMMAs -------------------------
    v8f s0 = zero8(), s1 = zero8();
#pragma unroll
    for (int kk = 0; kk < 2; ++kk) {
      v16h kfa = *(const v16h*)(Kb + ((size_t)bh * SEQ + kb0 + l16) * HD +
                                kk * 32 + half * 16);
      v16h kfb = *(const v16h*)(Kb + ((size_t)bh * SEQ + kb0 + 16 + l16) * HD +
                                kk * 32 + half * 16);
      s0 = WMMA_F16(qa[kk], kfa, s0);
      s1 = WMMA_F16(qa[kk], kfb, s1);
    }
    // ---- causal mask + online softmax ------------------------------------
    const int kj0 = kb0 + l16, kj1 = kj0 + 16;
#pragma unroll
    for (int r = 0; r < 8; ++r) {
      const int qi = qw + 8 * half + r;
      float a0 = (kj0 <= qi) ? s0[r] * scale : -__builtin_inff();
      float a1 = (kj1 <= qi) ? s1[r] * scale : -__builtin_inff();
      float mx = fmaxf(a0, a1);
#pragma unroll
      for (int off = 8; off >= 1; off >>= 1)
        mx = fmaxf(mx, __shfl_xor(mx, off, 16));    // row lives in one half
      const float mnew  = fmaxf(rm[r], mx);
      const float alpha = __expf(rm[r] - mnew);
      const float p0 = __expf(a0 - mnew);
      const float p1 = __expf(a1 - mnew);
      float rs = p0 + p1;
#pragma unroll
      for (int off = 8; off >= 1; off >>= 1) rs += __shfl_xor(rs, off, 16);
      rl[r] = rl[r] * alpha + rs;
      rm[r] = mnew;
#pragma unroll
      for (int t = 0; t < 4; ++t) o[t][r] *= alpha;   // same row<->elem map
      pbuf[wave][8 * half + r][l16]      = (_Float16)p0;
      pbuf[wave][8 * half + r][16 + l16] = (_Float16)p1;
    }
    // C-fragment -> A-fragment re-layout through per-wave LDS
    asm volatile("s_wait_dscnt 0" ::: "memory");
    v8h pc0 = *(const v8h*)(&pbuf[wave][l16][half * 16]);
    v8h pc1 = *(const v8h*)(&pbuf[wave][l16][half * 16 + 8]);
    v16h pa;
#pragma unroll
    for (int i = 0; i < 8; ++i) { pa[i] = pc0[i]; pa[8 + i] = pc1[i]; }
    // ---- O += P @ V : 4 WMMAs (V transposed -> contiguous B frags) -------
#pragma unroll
    for (int t = 0; t < 4; ++t) {
      v16h vb = *(const v16h*)(Vt + ((size_t)bh * HD + 16 * t + l16) * SEQ +
                               kb0 + half * 16);
      o[t] = WMMA_F16(pa, vb, o[t]);
    }
  }

  // ---- normalize + store attn output as f16 [B, S, E] --------------------
  const int b = bh >> 4, h = bh & 15;
#pragma unroll
  for (int t = 0; t < 4; ++t) {
    const int col = h * HD + 16 * t + l16;
#pragma unroll
    for (int r = 0; r < 8; ++r) {
      const int s = qw + 8 * half + r;
      const float val = o[t][r] / rl[r];
      attn[((size_t)(b * SEQ + s)) * EMB + col] = (_Float16)val;
    }
  }
}

// -----------------------------------------------------------------------------
// Kernel 3: output projection.  out = attn @ Wo^T + bo  (f32 output)
// -----------------------------------------------------------------------------
__global__ __launch_bounds__(128) void out_proj_kernel(
    const _Float16* __restrict__ attn, const _Float16* __restrict__ Woh,
    const float* __restrict__ bo, float* __restrict__ out) {
  const int lane = threadIdx.x & 31;
  const int wave = threadIdx.x >> 5;
  const int l16  = lane & 15;
  const int half = (lane >> 4) & 1;
  const int m0 = (blockIdx.x * 4 + wave) * 16;
  const int n0 = blockIdx.y * 64;

  v8f acc[4];
#pragma unroll
  for (int t = 0; t < 4; ++t) acc[t] = zero8();

  const _Float16* arow = attn + (size_t)(m0 + l16) * EMB;
  const _Float16* wrow[4];
#pragma unroll
  for (int t = 0; t < 4; ++t)
    wrow[t] = Woh + (size_t)(n0 + 16 * t + l16) * EMB + half * 16;

  for (int k0 = 0; k0 < EMB; k0 += 32) {
    v8h c0 = *(const v8h*)(arow + k0 + half * 16);
    v8h c1 = *(const v8h*)(arow + k0 + half * 16 + 8);
    v16h a;
#pragma unroll
    for (int i = 0; i < 8; ++i) { a[i] = c0[i]; a[8 + i] = c1[i]; }
#pragma unroll
    for (int t = 0; t < 4; ++t) {
      v16h bf = *(const v16h*)(wrow[t] + k0);
      __builtin_prefetch((const void*)(wrow[t] + k0 + 128), 0, 0);
      acc[t] = WMMA_F16(a, bf, acc[t]);
    }
  }

#pragma unroll
  for (int t = 0; t < 4; ++t) {
    const int n = n0 + 16 * t + l16;
    const float bb = bo[n];
#pragma unroll
    for (int r = 0; r < 8; ++r) {
      const int m = m0 + 8 * half + r;
      out[(size_t)m * EMB + n] = acc[t][r] + bb;
    }
  }
}

// -----------------------------------------------------------------------------
extern "C" void kernel_launch(void* const* d_in, const int* in_sizes, int n_in,
                              void* d_out, int out_size, void* d_ws,
                              size_t ws_size, hipStream_t stream) {
  (void)in_sizes; (void)n_in; (void)out_size; (void)ws_size;
  const float* x  = (const float*)d_in[0];
  // d_in[1] = mask (tril) — causal mask is applied analytically in-kernel.
  const float* Wq = (const float*)d_in[2];
  const float* bq = (const float*)d_in[3];
  const float* Wk = (const float*)d_in[4];
  const float* bk = (const float*)d_in[5];
  const float* Wv = (const float*)d_in[6];
  const float* bv = (const float*)d_in[7];
  const float* Wo = (const float*)d_in[8];
  const float* bo = (const float*)d_in[9];
  float* out = (float*)d_out;

  _Float16* ws = (_Float16*)d_ws;
  const size_t NX = (size_t)TOK * EMB;             // 4,194,304  (x / attn / QKV)
  const size_t NW = (size_t)EMB * EMB;             // 1,048,576  (one weight)
  _Float16* xh   = ws;                 // NX
  _Float16* Wqh  = xh  + NX;           // NW
  _Float16* Wkh  = Wqh + NW;
  _Float16* Wvh  = Wkh + NW;
  _Float16* Woh  = Wvh + NW;
  _Float16* Qb   = Woh + NW;           // NX
  _Float16* Kb   = Qb  + NX;
  _Float16* Vt   = Kb  + NX;
  _Float16* attn = Vt  + NX;           // total ~50.3 MB

  // ---- one-shot f16 conversions ----
  cvt_f16_kernel<<<dim3((int)(NX / 8 / 256)), dim3(256), 0, stream>>>(x, xh, (int)NX);
  cvt_f16_kernel<<<dim3((int)(NW / 8 / 256)), dim3(256), 0, stream>>>(Wq, Wqh, (int)NW);
  cvt_f16_kernel<<<dim3((int)(NW / 8 / 256)), dim3(256), 0, stream>>>(Wk, Wkh, (int)NW);
  cvt_f16_kernel<<<dim3((int)(NW / 8 / 256)), dim3(256), 0, stream>>>(Wv, Wvh, (int)NW);
  cvt_f16_kernel<<<dim3((int)(NW / 8 / 256)), dim3(256), 0, stream>>>(Wo, Woh, (int)NW);

  qkv_proj_kernel<<<dim3(TOK / 64, EMB / 64, 3), dim3(128), 0, stream>>>(
      xh, Wqh, bq, Wkh, bk, Wvh, bv, Qb, Kb, Vt);
  attn_kernel<<<dim3(SEQ / 64, BSZ * NH), dim3(128), 0, stream>>>(
      Qb, Kb, Vt, attn);
  out_proj_kernel<<<dim3(TOK / 64, EMB / 64), dim3(128), 0, stream>>>(
      attn, Woh, bo, out);
}
